// GraphormerEncoderLayer_15779709846384
// MI455X (gfx1250) — matrix operand order repaired and checked by reading, not compiled, not run on hardware
//
#include <hip/hip_runtime.h>
#include <hip/hip_bf16.h>
#include <math.h>

typedef __attribute__((ext_vector_type(16))) _Float16 v16h;
typedef __attribute__((ext_vector_type(8)))  _Float16 v8h;
typedef __attribute__((ext_vector_type(8)))  float    v8f;

#define TB 256
#define PA 40   // LDS pitch in halves for 32-wide K tiles (pad to dodge bank conflicts)

// ---------------------------------------------------------------------------
// WMMA fragment loads from LDS (layouts per CDNA5 ISA §7.12.2)
__device__ __forceinline__ v16h frag_a(const _Float16* lds, int rowBase, int lane) {
  int row = rowBase + (lane & 15);
  int kb  = (lane < 16) ? 0 : 8;
  const _Float16* p = lds + row * PA + kb;
  v8h lo = *(const v8h*)(p);
  v8h hi = *(const v8h*)(p + 16);
  v16h r;
#pragma unroll
  for (int i = 0; i < 8; ++i) { r[i] = lo[i]; r[i + 8] = hi[i]; }
  return r;
}

__device__ __forceinline__ v16h frag_b(const _Float16* lds, int colBase, int lane) {
  int col = colBase + (lane & 15);
  int kb  = (lane < 16) ? 0 : 8;
  const _Float16* p = lds + col * PA + kb;
  v8h lo = *(const v8h*)(p);
  v8h hi = *(const v8h*)(p + 16);
  v16h r;
#pragma unroll
  for (int i = 0; i < 8; ++i) { r[i] = lo[i]; r[i + 8] = hi[i]; }
  return r;
}

__device__ __forceinline__ v8f wmma_f16(v16h a, v16h b, v8f c) {
  return __builtin_amdgcn_wmma_f32_16x16x32_f16(false, a, false, b, (short)0, c,
                                                false, false);
}

__device__ __forceinline__ float gelu_tanh(float x) {
  const float c = 0.7978845608028654f;  // sqrt(2/pi)
  return 0.5f * x * (1.0f + tanhf(c * (x + 0.044715f * x * x * x)));
}

__device__ __forceinline__ float mishf(float x) {
  float sp = (x > 20.f) ? x : log1pf(__expf(x));
  return x * tanhf(sp);
}

// ---- wave32 + 8-wave block reductions (2 barriers each) --------------------
__device__ __forceinline__ float blockMaxF(float v, float* s8, int tid) {
#pragma unroll
  for (int o = 16; o > 0; o >>= 1) v = fmaxf(v, __shfl_xor(v, o, 32));
  if ((tid & 31) == 0) s8[tid >> 5] = v;
  __syncthreads();
  float r = s8[0];
#pragma unroll
  for (int i = 1; i < 8; ++i) r = fmaxf(r, s8[i]);
  __syncthreads();
  return r;
}

__device__ __forceinline__ float blockSumF(float v, float* s8, int tid) {
#pragma unroll
  for (int o = 16; o > 0; o >>= 1) v += __shfl_xor(v, o, 32);
  if ((tid & 31) == 0) s8[tid >> 5] = v;
  __syncthreads();
  float r = s8[0];
#pragma unroll
  for (int i = 1; i < 8; ++i) r += s8[i];
  __syncthreads();
  return r;
}

__device__ __forceinline__ unsigned blockAndU(unsigned v, unsigned* s8, int tid) {
#pragma unroll
  for (int o = 16; o > 0; o >>= 1) v &= (unsigned)__shfl_xor((int)v, o, 32);
  if ((tid & 31) == 0) s8[tid >> 5] = v;
  __syncthreads();
  unsigned r = s8[0];
#pragma unroll
  for (int i = 1; i < 8; ++i) r &= s8[i];
  __syncthreads();
  return r;
}

// ---------------------------------------------------------------------------
// Generic dense GEMM: C[M x Nout] = act(A[M x K] @ Wt[Nout x K]^T + bias) + resid
// Block tile 128x128, 8 waves (2M x 4N), wave 64x32 = 4x2 WMMA tiles.
// Register prefetch + LDS double buffering: one barrier per K step.
__global__ __launch_bounds__(TB)
void gemm_f16_wmma(const _Float16* __restrict__ A, int lda,
                   const _Float16* __restrict__ Wt, int ldw,
                   const float* __restrict__ bias,
                   const float* __restrict__ resid,
                   float* __restrict__ outF,
                   _Float16* __restrict__ outH,
                   int M, int Nout, int K, int act) {
  __shared__ __align__(16) _Float16 ldsA[2][128 * PA];
  __shared__ __align__(16) _Float16 ldsB[2][128 * PA];
  const int tid  = threadIdx.x;
  const int lane = tid & 31, wave = tid >> 5;
  const int wm = (wave & 1) * 64;
  const int wn = (wave >> 1) * 32;
  const int mBlock = blockIdx.y * 128;
  const int nBlock = blockIdx.x * 128;

  // straight-line staging: each thread owns 2 fixed 16B chunks per matrix
  const int r0 = tid >> 2;           // 0..63
  const int ck = (tid & 3) << 3;     // 0,8,16,24
  const int st0 = r0 * PA + ck;      // LDS store offsets (halves)
  const int st1 = st0 + 64 * PA;
  const _Float16* pa0 = A  + (size_t)(mBlock + r0) * lda + ck;
  const _Float16* pa1 = pa0 + (size_t)64 * lda;
  const _Float16* pb0 = Wt + (size_t)(nBlock + r0) * ldw + ck;
  const _Float16* pb1 = pb0 + (size_t)64 * ldw;

  _Float16* ca = &ldsA[0][0]; _Float16* na = &ldsA[1][0];
  _Float16* cb = &ldsB[0][0]; _Float16* nb = &ldsB[1][0];

  v8h ra0 = *(const v8h*)pa0, ra1 = *(const v8h*)pa1;
  v8h rb0 = *(const v8h*)pb0, rb1 = *(const v8h*)pb1;
  *(v8h*)(ca + st0) = ra0; *(v8h*)(ca + st1) = ra1;
  *(v8h*)(cb + st0) = rb0; *(v8h*)(cb + st1) = rb1;
  __syncthreads();

  v8f acc[4][2];
#pragma unroll
  for (int i = 0; i < 4; ++i)
#pragma unroll
    for (int j = 0; j < 2; ++j)
#pragma unroll
      for (int r = 0; r < 8; ++r) acc[i][j][r] = 0.f;

  for (int k0 = 0; k0 < K; k0 += 32) {
    const bool more = (k0 + 32 < K);
    if (more) {  // prefetch next K tile into registers
      pa0 += 32; pa1 += 32; pb0 += 32; pb1 += 32;
      ra0 = *(const v8h*)pa0; ra1 = *(const v8h*)pa1;
      rb0 = *(const v8h*)pb0; rb1 = *(const v8h*)pb1;
    }
    v16h af[4], bf[2];
#pragma unroll
    for (int mi = 0; mi < 4; ++mi) af[mi] = frag_a(ca, wm + mi * 16, lane);
#pragma unroll
    for (int ni = 0; ni < 2; ++ni) bf[ni] = frag_b(cb, wn + ni * 16, lane);
#pragma unroll
    for (int mi = 0; mi < 4; ++mi)
#pragma unroll
      for (int ni = 0; ni < 2; ++ni)
        acc[mi][ni] = wmma_f16(af[mi], bf[ni], acc[mi][ni]);
    if (more) {  // stage prefetched tile into the other buffer
      *(v8h*)(na + st0) = ra0; *(v8h*)(na + st1) = ra1;
      *(v8h*)(nb + st0) = rb0; *(v8h*)(nb + st1) = rb1;
    }
    __syncthreads();
    _Float16* t;
    t = ca; ca = na; na = t;
    t = cb; cb = nb; nb = t;
  }

  const int nCol = lane & 15;
  const int rOff = (lane >> 4) << 3;
#pragma unroll
  for (int mi = 0; mi < 4; ++mi)
#pragma unroll
    for (int ni = 0; ni < 2; ++ni) {
      int n  = nBlock + wn + ni * 16 + nCol;
      int mb = mBlock + wm + mi * 16 + rOff;
      float bvn = bias ? bias[n] : 0.f;
#pragma unroll
      for (int r = 0; r < 8; ++r) {
        int m = mb + r;
        float v = acc[mi][ni][r] + bvn;
        if (act == 1) v = gelu_tanh(v);
        if (resid) v += resid[(size_t)m * Nout + n];
        if (outF) outF[(size_t)m * Nout + n] = v;
        if (outH) outH[(size_t)m * Nout + n] = (_Float16)v;
      }
    }
}

// ---------------------------------------------------------------------------
// Attention scores per (b,g): S[n,m] = sum_d q[b,n,g,d]*k[b,m,g,d]
// M=N=512, K=D=96.  grid (4,4,B*G).
__global__ __launch_bounds__(TB)
void scores_wmma(const _Float16* __restrict__ q, const _Float16* __restrict__ k,
                 _Float16* __restrict__ S) {
  constexpr int Nn = 512, Gg = 8, Dd = 96, RS = Gg * Dd;
  __shared__ __align__(16) _Float16 ldsA[2][128 * PA];
  __shared__ __align__(16) _Float16 ldsB[2][128 * PA];
  const int tid = threadIdx.x, lane = tid & 31, wave = tid >> 5;
  const int wm = (wave & 1) * 64, wn = (wave >> 1) * 32;
  const int mBlock = blockIdx.y * 128, nBlock = blockIdx.x * 128;
  const int z = blockIdx.z;
  const int b = z >> 3, g = z & 7;
  const size_t baseQ = ((size_t)b * Nn * Gg + g) * Dd;
  const size_t baseS = (size_t)z * Nn * Nn;

  const int r0 = tid >> 2;
  const int ck = (tid & 3) << 3;
  const int st0 = r0 * PA + ck;
  const int st1 = st0 + 64 * PA;
  const _Float16* pa0 = q + baseQ + (size_t)(mBlock + r0) * RS + ck;
  const _Float16* pa1 = pa0 + (size_t)64 * RS;
  const _Float16* pb0 = k + baseQ + (size_t)(nBlock + r0) * RS + ck;
  const _Float16* pb1 = pb0 + (size_t)64 * RS;

  _Float16* ca = &ldsA[0][0]; _Float16* na = &ldsA[1][0];
  _Float16* cb = &ldsB[0][0]; _Float16* nb = &ldsB[1][0];

  v8h ra0 = *(const v8h*)pa0, ra1 = *(const v8h*)pa1;
  v8h rb0 = *(const v8h*)pb0, rb1 = *(const v8h*)pb1;
  *(v8h*)(ca + st0) = ra0; *(v8h*)(ca + st1) = ra1;
  *(v8h*)(cb + st0) = rb0; *(v8h*)(cb + st1) = rb1;
  __syncthreads();

  v8f acc[4][2];
#pragma unroll
  for (int i = 0; i < 4; ++i)
#pragma unroll
    for (int j = 0; j < 2; ++j)
#pragma unroll
      for (int r = 0; r < 8; ++r) acc[i][j][r] = 0.f;

#pragma unroll
  for (int k0 = 0; k0 < Dd; k0 += 32) {
    const bool more = (k0 + 32 < Dd);
    if (more) {
      pa0 += 32; pa1 += 32; pb0 += 32; pb1 += 32;
      ra0 = *(const v8h*)pa0; ra1 = *(const v8h*)pa1;
      rb0 = *(const v8h*)pb0; rb1 = *(const v8h*)pb1;
    }
    v16h af[4], bf[2];
#pragma unroll
    for (int mi = 0; mi < 4; ++mi) af[mi] = frag_a(ca, wm + mi * 16, lane);
#pragma unroll
    for (int ni = 0; ni < 2; ++ni) bf[ni] = frag_b(cb, wn + ni * 16, lane);
#pragma unroll
    for (int mi = 0; mi < 4; ++mi)
#pragma unroll
      for (int ni = 0; ni < 2; ++ni)
        acc[mi][ni] = wmma_f16(af[mi], bf[ni], acc[mi][ni]);
    if (more) {
      *(v8h*)(na + st0) = ra0; *(v8h*)(na + st1) = ra1;
      *(v8h*)(nb + st0) = rb0; *(v8h*)(nb + st1) = rb1;
    }
    __syncthreads();
    _Float16* t;
    t = ca; ca = na; na = t;
    t = cb; cb = nb; nb = t;
  }

  const int nCol = lane & 15;
  const int rOff = (lane >> 4) << 3;
#pragma unroll
  for (int mi = 0; mi < 4; ++mi)
#pragma unroll
    for (int ni = 0; ni < 2; ++ni) {
      int n  = nBlock + wn + ni * 16 + nCol;
      int mb = mBlock + wm + mi * 16 + rOff;
#pragma unroll
      for (int r = 0; r < 8; ++r)
        S[baseS + (size_t)(mb + r) * Nn + n] = (_Float16)acc[mi][ni][r];
    }
}

// ---------------------------------------------------------------------------
// Admixture + mish + bias + softmax.  One block per (b,n): loads the 8 per-g
// score/eps rows ONCE and produces all 16 l-rows (16x less eps/bias traffic).
__global__ __launch_bounds__(TB)
void mix_softmax(const _Float16* __restrict__ S, const float* __restrict__ eps,
                 const float* __restrict__ ebias, const float* __restrict__ sigma,
                 const float* __restrict__ p, _Float16* __restrict__ Aout) {
  constexpr int Nn = 512, Gg = 8, Ll = 16;
  __shared__ float red8[8];
  __shared__ unsigned ired8[8];
  __shared__ float wsh[Gg * Ll];
  const int tid = threadIdx.x;
  const int n = blockIdx.x, b = blockIdx.y;
  const int m0 = tid, m1 = tid + TB;

  if (tid < Gg * Ll) wsh[tid] = p[tid];

  float mv0[Gg], mv1[Gg];
  unsigned zf = 0xFFu;
#pragma unroll
  for (int g = 0; g < Gg; ++g) {
    size_t off = (((size_t)b * Gg + g) * Nn + n) * Nn;
    float s0 = (float)S[off + m0], s1 = (float)S[off + m1];
    float sg = sigma[g];
    float s2 = sg * sg;
    mv0[g] = s0 + s2 * eps[off + m0];
    mv1[g] = s1 + s2 * eps[off + m1];
    if (s0 != 0.f || s1 != 0.f) zf &= ~(1u << g);
  }
  unsigned zall = blockAndU(zf, ired8, tid);  // pad1: zero score row => drop g
#pragma unroll
  for (int g = 0; g < Gg; ++g)
    if ((zall >> g) & 1u) { mv0[g] = 0.f; mv1[g] = 0.f; }

  size_t bOff = ((size_t)b * Nn + n) * Nn;
  float eb0 = ebias[bOff + m0], eb1 = ebias[bOff + m1];
  const float scale = 0.03608439182435161f;  // 768^-0.5

  for (int l = 0; l < Ll; ++l) {
    float a0 = 0.f, a1 = 0.f;
#pragma unroll
    for (int g = 0; g < Gg; ++g) {
      float w = wsh[g * Ll + l];
      a0 += w * mv0[g];
      a1 += w * mv1[g];
    }
    unsigned z2 = blockAndU((a0 == 0.f && a1 == 0.f) ? 1u : 0u, ired8, tid);
    size_t outOff = (((size_t)b * Ll + l) * Nn + n) * Nn;
    if (z2 != 0u) {  // pad2: -inf row -> softmax NaN -> nan_to_num -> 0
      Aout[outOff + m0] = (_Float16)0.f;
      Aout[outOff + m1] = (_Float16)0.f;
      continue;
    }
    float y0 = mishf(a0) * scale + eb0;
    float y1 = mishf(a1) * scale + eb1;
    float mx = blockMaxF(fmaxf(y0, y1), red8, tid);
    float e0 = __expf(y0 - mx), e1 = __expf(y1 - mx);
    float sum = blockSumF(e0 + e1, red8, tid);
    float inv = (sum > 0.f) ? 1.f / sum : 0.f;
    Aout[outOff + m0] = (_Float16)(e0 * inv);
    Aout[outOff + m1] = (_Float16)(e1 * inv);
  }
}

// ---------------------------------------------------------------------------
// O[b,n,l,:] = Attn[b,l] (512x512) @ V_l (512x96).  Block tile 128(M)x96(N),
// 8 waves as 4(M)x2(N), wave 32x48 = 2x3 WMMA tiles.  grid (1,4,B*L).
__global__ __launch_bounds__(TB)
void av_wmma(const _Float16* __restrict__ Attn, const _Float16* __restrict__ vv,
             _Float16* __restrict__ O) {
  constexpr int Nn = 512, Ll = 16, Dd = 96, LD = Ll * Dd;
  __shared__ __align__(16) _Float16 ldsA[2][128 * PA];
  __shared__ __align__(16) _Float16 ldsB[2][96 * PA];
  const int tid = threadIdx.x, lane = tid & 31, wave = tid >> 5;
  const int wm = (wave & 3) * 32;
  const int wn = (wave >> 2) * 48;
  const int mBlock = blockIdx.y * 128;
  const int z = blockIdx.z;
  const int b = z >> 4, l = z & 15;
  const size_t baseA = (size_t)z * Nn * Nn;
  const size_t baseV = (size_t)b * Nn * LD + (size_t)l * Dd;  // + m*LD + d

  const int r0 = tid >> 2;
  const int ck = (tid & 3) << 3;
  const int st0 = r0 * PA + ck;
  const int st1 = st0 + 64 * PA;
  const _Float16* pa0 = Attn + baseA + (size_t)(mBlock + r0) * Nn + ck;
  const _Float16* pa1 = pa0 + (size_t)64 * Nn;

  // v-tile transpose offsets: 96(d) x 32(m) per K step, 12 halves per thread
  int voff[12], loff[12];
#pragma unroll
  for (int j = 0; j < 12; ++j) {
    int e = tid + j * TB;
    int d = e % 96, m = e / 96;
    voff[j] = m * LD + d;
    loff[j] = d * PA + m;
  }

  _Float16* ca = &ldsA[0][0]; _Float16* na = &ldsA[1][0];
  _Float16* cb = &ldsB[0][0]; _Float16* nb = &ldsB[1][0];

  v8h ra0 = *(const v8h*)pa0, ra1 = *(const v8h*)pa1;
  _Float16 rv[12];
#pragma unroll
  for (int j = 0; j < 12; ++j) rv[j] = vv[baseV + voff[j]];
  *(v8h*)(ca + st0) = ra0; *(v8h*)(ca + st1) = ra1;
#pragma unroll
  for (int j = 0; j < 12; ++j) cb[loff[j]] = rv[j];
  __syncthreads();

  v8f acc[2][3];
#pragma unroll
  for (int i = 0; i < 2; ++i)
#pragma unroll
    for (int j = 0; j < 3; ++j)
#pragma unroll
      for (int r = 0; r < 8; ++r) acc[i][j][r] = 0.f;

  for (int k0 = 0; k0 < Nn; k0 += 32) {
    const bool more = (k0 + 32 < Nn);
    if (more) {
      pa0 += 32; pa1 += 32;
      ra0 = *(const v8h*)pa0; ra1 = *(const v8h*)pa1;
      size_t vb = baseV + (size_t)(k0 + 32) * LD;
#pragma unroll
      for (int j = 0; j < 12; ++j) rv[j] = vv[vb + voff[j]];
    }
    v16h af[2], bf[3];
#pragma unroll
    for (int mi = 0; mi < 2; ++mi) af[mi] = frag_a(ca, wm + mi * 16, lane);
#pragma unroll
    for (int ni = 0; ni < 3; ++ni) bf[ni] = frag_b(cb, wn + ni * 16, lane);
#pragma unroll
    for (int mi = 0; mi < 2; ++mi)
#pragma unroll
      for (int ni = 0; ni < 3; ++ni)
        acc[mi][ni] = wmma_f16(af[mi], bf[ni], acc[mi][ni]);
    if (more) {
      *(v8h*)(na + st0) = ra0; *(v8h*)(na + st1) = ra1;
#pragma unroll
      for (int j = 0; j < 12; ++j) nb[loff[j]] = rv[j];
    }
    __syncthreads();
    _Float16* t;
    t = ca; ca = na; na = t;
    t = cb; cb = nb; nb = t;
  }

  const int nCol = lane & 15;
  const int rOff = (lane >> 4) << 3;
#pragma unroll
  for (int mi = 0; mi < 2; ++mi)
#pragma unroll
    for (int ni = 0; ni < 3; ++ni) {
      int d  = wn + ni * 16 + nCol;
      int mb = mBlock + wm + mi * 16 + rOff;
#pragma unroll
      for (int r = 0; r < 8; ++r)
        O[baseV + (size_t)(mb + r) * LD + d] = (_Float16)acc[mi][ni][r];
    }
}

// ---------------------------------------------------------------------------
// LayerNorm over H=768, one block per row; writes f16 for WMMA consumption.
__global__ __launch_bounds__(TB)
void layernorm_f16(const float* __restrict__ x, const float* __restrict__ gam,
                   const float* __restrict__ bet, _Float16* __restrict__ out) {
  constexpr int Hh = 768;
  __shared__ float red8[8];
  const int row = blockIdx.x, tid = threadIdx.x;
  const float* xr = x + (size_t)row * Hh;
  float v0 = xr[tid], v1 = xr[tid + 256], v2 = xr[tid + 512];
  float mean = blockSumF(v0 + v1 + v2, red8, tid) * (1.f / Hh);
  float d0 = v0 - mean, d1 = v1 - mean, d2 = v2 - mean;
  float var = blockSumF(d0 * d0 + d1 * d1 + d2 * d2, red8, tid) * (1.f / Hh);
  float inv = rsqrtf(var + 1e-5f);
  _Float16* orow = out + (size_t)row * Hh;
  orow[tid]       = (_Float16)(d0 * inv * gam[tid] + bet[tid]);
  orow[tid + 256] = (_Float16)(d1 * inv * gam[tid + 256] + bet[tid + 256]);
  orow[tid + 512] = (_Float16)(d2 * inv * gam[tid + 512] + bet[tid + 512]);
}

__global__ void cvt_f32_f16(const float* __restrict__ in,
                            _Float16* __restrict__ out, int n) {
  int i = blockIdx.x * blockDim.x + threadIdx.x;
  if (i < n) out[i] = (_Float16)in[i];
}

// ---------------------------------------------------------------------------
extern "C" void kernel_launch(void* const* d_in, const int* in_sizes, int n_in,
                              void* d_out, int out_size, void* d_ws,
                              size_t ws_size, hipStream_t stream) {
  (void)in_sizes; (void)n_in; (void)out_size; (void)ws_size;
  const float* x     = (const float*)d_in[0];
  const float* ebias = (const float*)d_in[1];
  const float* eps   = (const float*)d_in[2];
  const float* Wq    = (const float*)d_in[3];
  const float* Wk    = (const float*)d_in[4];
  const float* Wv    = (const float*)d_in[5];
  const float* bv    = (const float*)d_in[6];
  const float* sigma = (const float*)d_in[7];
  const float* p     = (const float*)d_in[8];
  const float* Wo    = (const float*)d_in[9];
  const float* ln1g  = (const float*)d_in[10];
  const float* ln1b  = (const float*)d_in[11];
  const float* ln2g  = (const float*)d_in[12];
  const float* ln2b  = (const float*)d_in[13];
  const float* W1    = (const float*)d_in[14];
  const float* b1    = (const float*)d_in[15];
  const float* W2    = (const float*)d_in[16];
  const float* b2    = (const float*)d_in[17];
  float* out = (float*)d_out;

  constexpr int B = 16, N = 512, H = 768, G = 8, L = 16, FFN = 3072, D = 96;
  constexpr int LD = L * D;
  const int M = B * N;

  size_t off = 0;
  auto alloc = [&](size_t elems, size_t esz) -> void* {
    void* ptr = (char*)d_ws + off;
    off += (elems * esz + 255) & ~(size_t)255;
    return ptr;
  };
  _Float16* h16  = (_Float16*)alloc((size_t)M * H, 2);
  _Float16* q16  = (_Float16*)alloc((size_t)M * H, 2);
  _Float16* k16  = (_Float16*)alloc((size_t)M * H, 2);
  _Float16* v16  = (_Float16*)alloc((size_t)M * LD, 2);
  _Float16* Wq16 = (_Float16*)alloc((size_t)H * H, 2);
  _Float16* Wk16 = (_Float16*)alloc((size_t)H * H, 2);
  _Float16* Wv16 = (_Float16*)alloc((size_t)LD * H, 2);
  _Float16* Wo16 = (_Float16*)alloc((size_t)H * LD, 2);
  _Float16* W116 = (_Float16*)alloc((size_t)FFN * H, 2);
  _Float16* W216 = (_Float16*)alloc((size_t)H * FFN, 2);
  _Float16* S16  = (_Float16*)alloc((size_t)B * G * N * N, 2);
  _Float16* A16  = (_Float16*)alloc((size_t)B * L * N * N, 2);
  _Float16* O16  = (_Float16*)alloc((size_t)M * LD, 2);
  float*    x1   = (float*)alloc((size_t)M * H, 4);
  _Float16* h216 = (_Float16*)alloc((size_t)M * H, 2);
  _Float16* t16  = (_Float16*)alloc((size_t)M * FFN, 2);

  auto cvt = [&](const float* src, _Float16* dst, size_t n) {
    int blocks = (int)((n + 255) / 256);
    cvt_f32_f16<<<blocks, 256, 0, stream>>>(src, dst, (int)n);
  };
  cvt(Wq, Wq16, (size_t)H * H);
  cvt(Wk, Wk16, (size_t)H * H);
  cvt(Wv, Wv16, (size_t)LD * H);
  cvt(Wo, Wo16, (size_t)H * LD);
  cvt(W1, W116, (size_t)FFN * H);
  cvt(W2, W216, (size_t)H * FFN);

  // h = LN1(x)
  layernorm_f16<<<M, TB, 0, stream>>>(x, ln1g, ln1b, h16);

  // q/k/v projections
  gemm_f16_wmma<<<dim3(H / 128, M / 128), TB, 0, stream>>>(
      h16, H, Wq16, H, nullptr, nullptr, nullptr, q16, M, H, H, 0);
  gemm_f16_wmma<<<dim3(H / 128, M / 128), TB, 0, stream>>>(
      h16, H, Wk16, H, nullptr, nullptr, nullptr, k16, M, H, H, 0);
  gemm_f16_wmma<<<dim3(LD / 128, M / 128), TB, 0, stream>>>(
      h16, H, Wv16, H, bv, nullptr, nullptr, v16, M, LD, H, 0);

  // per-(b,g) score matrices
  scores_wmma<<<dim3(4, 4, B * G), TB, 0, stream>>>(q16, k16, S16);

  // admixture + mish + bias + softmax -> f16 probabilities (one block per (b,n))
  mix_softmax<<<dim3(N, B), TB, 0, stream>>>(S16, eps, ebias, sigma, p, A16);

  // per-(b,l) O = A @ V_l
  av_wmma<<<dim3(1, 4, B * L), TB, 0, stream>>>(A16, v16, O16);

  // x1 = x + O @ Wo^T
  gemm_f16_wmma<<<dim3(H / 128, M / 128), TB, 0, stream>>>(
      O16, LD, Wo16, LD, nullptr, x, x1, nullptr, M, H, LD, 0);

  // h2 = LN2(x1)
  layernorm_f16<<<M, TB, 0, stream>>>(x1, ln2g, ln2b, h216);

  // FFN
  gemm_f16_wmma<<<dim3(FFN / 128, M / 128), TB, 0, stream>>>(
      h216, H, W116, H, b1, nullptr, nullptr, t16, M, FFN, H, 1);
  gemm_f16_wmma<<<dim3(H / 128, M / 128), TB, 0, stream>>>(
      t16, FFN, W216, FFN, b2, x1, out, nullptr, M, H, FFN, 0);
}
